// GeometryKernelAttention_25786983645930
// MI455X (gfx1250) — compile-verified, matrix operand order checked by reading, not run on hardware
//
#include <hip/hip_runtime.h>

// ---------------------------------------------------------------------------
// GeometryKernelAttention, fully fused for MI455X (gfx1250, wave32, WMMA).
// B=2 N=8192 K=16 C=256 H=8 HD=32.  One WG = 16 queries (=> 256 k/v rows).
// All intermediates in LDS (~295KB of the 320KB WGP LDS):
//   LN(q,k) -> q/k proj (WMMA f16, col-major D stores = ds_store_b128)
//   -> scores+softmax (vector-k VALU) -> LN(v) -> v proj (WMMA)
//   -> attn combine (vector-k) -> o proj (WMMA) -> fp32 out.
// ---------------------------------------------------------------------------

#define BB   2
#define NN   8192
#define KKN  16
#define CC   256
#define HH   8
#define HD   32
#define NT   16            // queries per workgroup
#define RK   (NT * KKN)    // 256 k/v rows per workgroup
#define LDSC (CC + 8)      // padded row stride (halves) for A-operand buffers
#define PSTR (RK + 8)      // padded row-count stride for column-major proj buf
#define NTHREADS 256
#define NWAVES   8

typedef __attribute__((ext_vector_type(16))) _Float16 v16h;
typedef __attribute__((ext_vector_type(8)))  _Float16 v8h;
typedef __attribute__((ext_vector_type(8)))  float    v8f;

__device__ __forceinline__ v16h cat8(v8h lo, v8h hi) {
  return __builtin_shufflevector(lo, hi, 0,1,2,3,4,5,6,7,8,9,10,11,12,13,14,15);
}

// A-fragment (16x32 f16) from LDS, row-major rows of stride LDSC.
// Lane m(0-15): K 0..7 & 16..23 ; lane m+16: K 8..15 & 24..31 (ISA 7.12.2).
__device__ __forceinline__ v16h load_frag_lds(const _Float16* base, int row0,
                                              int kc, int lane) {
  int m   = lane & 15;
  int klo = (lane >> 4) << 3;
  const _Float16* p = base + (size_t)(row0 + m) * LDSC + kc + klo;
  v8h lo = *(const v8h*)(p);
  v8h hi = *(const v8h*)(p + 16);
  return cat8(lo, hi);
}

// B-fragment (32x16 f16) for out = x @ W.T : column j of B == row j of W.
// W is pre-converted f16, row-major [256][256] in workspace (L2-resident).
__device__ __forceinline__ v16h load_frag_w(const _Float16* W, int j0,
                                            int kc, int lane) {
  int n   = lane & 15;
  int klo = (lane >> 4) << 3;
  const _Float16* p = W + (size_t)(j0 + n) * CC + kc + klo;
  v8h lo = *(const v8h*)(p);
  v8h hi = *(const v8h*)(p + 16);
  return cat8(lo, hi);
}

// LayerNorm nrows x 256 fp32 rows from global -> f16 rows in LDS (row-major).
// 16 groups of 16 lanes; each lane owns 16 fixed columns (g/b preloaded).
__device__ __forceinline__ void ln_rows(const float* __restrict__ src, int nrows,
                                        const float* __restrict__ gamma,
                                        const float* __restrict__ beta,
                                        _Float16* dst, int tid) {
  int g  = tid >> 4;
  int sl = tid & 15;
  float gg[16], bb[16];
#pragma unroll
  for (int i = 0; i < 16; ++i) {
    gg[i] = gamma[sl * 16 + i];
    bb[i] = beta [sl * 16 + i];
  }
  for (int r = g; r < nrows; r += 16) {
    const float* p = src + (size_t)r * CC + sl * 16;
    if (r + 16 < nrows) __builtin_prefetch(p + (size_t)16 * CC, 0, 0);
    float x[16], s = 0.f, s2 = 0.f;
#pragma unroll
    for (int i = 0; i < 16; ++i) {
      x[i] = p[i];
      s += x[i];
      s2 += x[i] * x[i];
    }
#pragma unroll
    for (int m = 8; m >= 1; m >>= 1) {
      s  += __shfl_xor(s,  m, 16);
      s2 += __shfl_xor(s2, m, 16);
    }
    float mu   = s * (1.f / CC);
    float var  = s2 * (1.f / CC) - mu * mu;
    float rstd = rsqrtf(var + 1e-5f);
    _Float16* d = dst + (size_t)r * LDSC + sl * 16;
#pragma unroll
    for (int i = 0; i < 16; ++i)
      d[i] = (_Float16)((x[i] - mu) * rstd * gg[i] + bb[i]);
  }
}

// GEMM: LDS A (nrows x 256 f16, row-major) x W.T (f16) + bias ->
// LDS f16 COLUMN-major: Out[col * rstr + row].  Each wave computes pairs of
// 16x16 col-tiles per row-tile so the A fragment is loaded once per 2 WMMAs,
// and each lane's 8 D values (one column, 8 consecutive rows) store as one
// ds_store_b128.
__device__ __forceinline__ void gemm_cm(const _Float16* A, int nrows,
                                        const _Float16* __restrict__ W,
                                        const float* __restrict__ bias,
                                        _Float16* Out, int rstr,
                                        int wave, int lane) {
  int npairs = (nrows >> 4) * 8;            // row-tiles x 8 col-tile-pairs
  for (int t = wave; t < npairs; t += NWAVES) {
    int rt = t >> 3, c0 = (t & 7) * 32;
    v8f acc0 = {}, acc1 = {};
#pragma unroll
    for (int kc = 0; kc < CC; kc += 32) {
      v16h a  = load_frag_lds(A, rt * 16, kc, lane);
      v16h b0 = load_frag_w(W, c0,      kc, lane);
      v16h b1 = load_frag_w(W, c0 + 16, kc, lane);
      acc0 = __builtin_amdgcn_wmma_f32_16x16x32_f16(false, a, false, b0,
                                                    (short)0, acc0, false, false);
      acc1 = __builtin_amdgcn_wmma_f32_16x16x32_f16(false, a, false, b1,
                                                    (short)0, acc1, false, false);
    }
    int mb = rt * 16 + ((lane >> 4) << 3);  // D layout: lanes16-31 hold M=r+8
#pragma unroll
    for (int s = 0; s < 2; ++s) {
      int   j  = c0 + s * 16 + (lane & 15);
      float bj = bias[j];
      v8f   ac = s ? acc1 : acc0;
      v8h   o;
#pragma unroll
      for (int r = 0; r < 8; ++r) o[r] = (_Float16)(ac[r] + bj);
      *(v8h*)(Out + (size_t)j * rstr + mb) = o;   // one b128 per tile per lane
    }
  }
}

__global__ void __launch_bounds__(NTHREADS)
geo_attn_fused(const float* __restrict__ query, const float* __restrict__ key,
               const float* __restrict__ value,
               const float* lnq_g, const float* lnq_b,
               const _Float16* __restrict__ Wq_h, const float* bq,
               const float* lnk_g, const float* lnk_b,
               const _Float16* __restrict__ Wk_h, const float* bk,
               const float* lnv_g, const float* lnv_b,
               const _Float16* __restrict__ Wv_h, const float* bv,
               const _Float16* __restrict__ Wo_h, const float* bo,
               float* __restrict__ out) {
  extern __shared__ _Float16 smem[];
  _Float16* Xn = smem;                    // 256 x 264 row-major: LN'd k, then v
  _Float16* P  = Xn + (size_t)RK * LDSC;  // 256 x 264 col-major: proj k, then v
  _Float16* Xq = P  + (size_t)CC * PSTR;  // 16 x 264 row-major: LN'd q, then attn out
  _Float16* Qp = Xq + (size_t)NT * LDSC;  // 256 x 16 col-major: projected q
  float*    S  = (float*)(Qp + (size_t)CC * NT);  // 16x8x16 softmax probs

  int tid  = threadIdx.x;
  int wave = tid >> 5;
  int lane = tid & 31;
  int b    = blockIdx.x / (NN / NT);
  int n0   = (blockIdx.x % (NN / NT)) * NT;

  const float* qsrc = query + ((size_t)b * NN + n0) * CC;
  const float* ksrc = key   + ((size_t)b * NN + n0) * KKN * CC; // rows contiguous
  const float* vsrc = value + ((size_t)b * NN + n0) * KKN * CC;

  // ---- stage 1: LayerNorm q and k into LDS (f16) ----
  ln_rows(qsrc, NT, lnq_g, lnq_b, Xq, tid);
  ln_rows(ksrc, RK, lnk_g, lnk_b, Xn, tid);
  __syncthreads();

  // ---- stage 2: q / k projections (WMMA), column-major D in LDS ----
  gemm_cm(Xq, NT, Wq_h, bq, Qp, NT,   wave, lane);
  gemm_cm(Xn, RK, Wk_h, bk, P,  PSTR, wave, lane);
  __syncthreads();

  // ---- stage 3: scores + softmax (one thread per (n,h); vector over k),
  //               plus LayerNorm(v) into freed Xn ----
  if (tid < NT * HH) {
    int n = tid >> 3, h = tid & 7;
    float sc[KKN];
#pragma unroll
    for (int k = 0; k < KKN; ++k) sc[k] = 0.f;
#pragma unroll
    for (int d = 0; d < HD; ++d) {
      int   c  = h * HD + d;
      float qv = (float)Qp[(size_t)c * NT + n];
      const v8h* kv = (const v8h*)(P + (size_t)c * PSTR + n * KKN);
      v8h k0 = kv[0], k1 = kv[1];
#pragma unroll
      for (int k = 0; k < 8; ++k) {
        sc[k]     += qv * (float)k0[k];
        sc[k + 8] += qv * (float)k1[k];
      }
    }
    float mx = sc[0] * 0.17677669529663687f;   // HD^-0.5
#pragma unroll
    for (int k = 0; k < KKN; ++k) {
      sc[k] *= 0.17677669529663687f;
      mx = fmaxf(mx, sc[k]);
    }
    float sum = 0.f;
#pragma unroll
    for (int k = 0; k < KKN; ++k) { sc[k] = __expf(sc[k] - mx); sum += sc[k]; }
    float inv = 1.f / sum;
    float* sp = S + ((size_t)n * HH + h) * KKN;
#pragma unroll
    for (int k = 0; k < KKN; ++k) sp[k] = sc[k] * inv;
  }
  ln_rows(vsrc, RK, lnv_g, lnv_b, Xn, tid);
  __syncthreads();

  // ---- stage 4: v projection (WMMA) overwrites P (col-major) ----
  gemm_cm(Xn, RK, Wv_h, bv, P, PSTR, wave, lane);
  __syncthreads();

  // ---- stage 5: attn-weighted combine -> Xq (row-major f16 A-operand) ----
  {
    int n  = tid >> 4;
    int c0 = (tid & 15) * 16;
    int h  = c0 >> 5;                  // constant over this thread's 16 columns
    const float* pr = S + ((size_t)n * HH + h) * KKN;
    float pk[KKN];
#pragma unroll
    for (int k = 0; k < KKN; ++k) pk[k] = pr[k];
#pragma unroll
    for (int i = 0; i < 16; ++i) {
      int c = c0 + i;
      const v8h* vv = (const v8h*)(P + (size_t)c * PSTR + n * KKN);
      v8h v0 = vv[0], v1 = vv[1];
      float acc = 0.f;
#pragma unroll
      for (int k = 0; k < 8; ++k)
        acc += pk[k] * (float)v0[k] + pk[k + 8] * (float)v1[k];
      Xq[(size_t)n * LDSC + c] = (_Float16)acc;
    }
  }
  __syncthreads();

  // ---- stage 6: output projection (WMMA) -> global fp32 ----
  {
    float* orow = out + ((size_t)b * NN + n0) * CC;
    for (int t = wave; t < 8; t += NWAVES) {   // 8 col-tile pairs, 1 per wave
      int c0 = t * 32;
      v8f acc0 = {}, acc1 = {};
#pragma unroll
      for (int kc = 0; kc < CC; kc += 32) {
        v16h a  = load_frag_lds(Xq, 0, kc, lane);
        v16h b0 = load_frag_w(Wo_h, c0,      kc, lane);
        v16h b1 = load_frag_w(Wo_h, c0 + 16, kc, lane);
        acc0 = __builtin_amdgcn_wmma_f32_16x16x32_f16(false, a, false, b0,
                                                      (short)0, acc0, false, false);
        acc1 = __builtin_amdgcn_wmma_f32_16x16x32_f16(false, a, false, b1,
                                                      (short)0, acc1, false, false);
      }
      int mb = (lane >> 4) << 3;
#pragma unroll
      for (int s = 0; s < 2; ++s) {
        int   j  = c0 + s * 16 + (lane & 15);
        float bj = bo[j];
        v8f   ac = s ? acc1 : acc0;
#pragma unroll
        for (int r = 0; r < 8; ++r)
          orow[(size_t)(mb + r) * CC + j] = ac[r] + bj;
      }
    }
  }
}

// Convert the four 256x256 fp32 weights to f16 in workspace (row-major).
__global__ void convert_weights(const float* __restrict__ Wq,
                                const float* __restrict__ Wk,
                                const float* __restrict__ Wv,
                                const float* __restrict__ Wo,
                                _Float16* __restrict__ dst) {
  int i = blockIdx.x * blockDim.x + threadIdx.x; // 0 .. 4*65536-1
  const float* srcs[4] = {Wq, Wk, Wv, Wo};
  int w = i >> 16;
  int e = i & 65535;
  dst[i] = (_Float16)srcs[w][e];
}

extern "C" void kernel_launch(void* const* d_in, const int* in_sizes, int n_in,
                              void* d_out, int out_size, void* d_ws, size_t ws_size,
                              hipStream_t stream) {
  (void)in_sizes; (void)n_in; (void)out_size; (void)ws_size;
  const float* query = (const float*)d_in[0];
  const float* key   = (const float*)d_in[1];
  const float* value = (const float*)d_in[2];
  const float* lnq_g = (const float*)d_in[3];
  const float* lnq_b = (const float*)d_in[4];
  const float* Wq    = (const float*)d_in[5];
  const float* bq    = (const float*)d_in[6];
  const float* lnk_g = (const float*)d_in[7];
  const float* lnk_b = (const float*)d_in[8];
  const float* Wk    = (const float*)d_in[9];
  const float* bk    = (const float*)d_in[10];
  const float* lnv_g = (const float*)d_in[11];
  const float* lnv_b = (const float*)d_in[12];
  const float* Wv    = (const float*)d_in[13];
  const float* bv    = (const float*)d_in[14];
  const float* Wo    = (const float*)d_in[15];
  const float* bo    = (const float*)d_in[16];
  float* out = (float*)d_out;

  _Float16* Wb = (_Float16*)d_ws;  // 4 * 65536 f16 = 512 KB
  convert_weights<<<(4 * 65536) / NTHREADS, NTHREADS, 0, stream>>>(Wq, Wk, Wv, Wo, Wb);

  const _Float16* Wq_h = Wb;
  const _Float16* Wk_h = Wb + 1 * 65536;
  const _Float16* Wv_h = Wb + 2 * 65536;
  const _Float16* Wo_h = Wb + 3 * 65536;

  size_t smem = (size_t)(RK * LDSC + CC * PSTR + NT * LDSC + CC * NT)
                    * sizeof(_Float16)
              + (size_t)NT * HH * KKN * sizeof(float);   // ~295 KB < 320 KB
  hipFuncSetAttribute((const void*)geo_attn_fused,
                      hipFuncAttributeMaxDynamicSharedMemorySize, (int)smem);

  geo_attn_fused<<<BB * (NN / NT), NTHREADS, smem, stream>>>(
      query, key, value,
      lnq_g, lnq_b, Wq_h, bq,
      lnk_g, lnk_b, Wk_h, bk,
      lnv_g, lnv_b, Wv_h, bv,
      Wo_h, bo, out);
}
